// TabR_26173530702530
// MI455X (gfx1250) — matrix operand order repaired and checked by reading, not compiled, not run on hardware
//
#include <hip/hip_runtime.h>
#include <hip/hip_bf16.h>

using bf16 = __bf16;
typedef __attribute__((ext_vector_type(16))) __bf16 v16bf;
typedef __attribute__((ext_vector_type(8)))  __bf16 v8bf;
typedef __attribute__((ext_vector_type(8)))  float  v8f;

#define NB  1024      // batch
#define NCAND 50000
#define CTX 96
#define DIN 128
#define DM  256
#define DB  512
#define NOUT 10
#define CCH 10000     // candidate encode chunk (div by 16)
#define QCH 256       // query context chunk

static __device__ __forceinline__ v8f wmma_bf16(v16bf a, v16bf b, v8f c) {
    return __builtin_amdgcn_wmma_f32_16x16x32_bf16(false, a, false, b, (short)0, c, false, false);
}

// Combine two contiguous 8-element bf16 chunks (at p and p+8) into an A fragment.
static __device__ __forceinline__ v16bf a_frag_at(const bf16* p) {
    v8bf lo = *(const v8bf*)p;
    v8bf hi = *(const v8bf*)(p + 8);
    return __builtin_shufflevector(lo, hi, 0,1,2,3,4,5,6,7,8,9,10,11,12,13,14,15);
}

// Load a 16x32 bf16 A-fragment from a row-major matrix (ld = K).
static __device__ __forceinline__ v16bf load_a_frag(const bf16* __restrict__ A, int ld,
                                                    int row, int kk, int lane) {
    return a_frag_at(A + (size_t)row * ld + kk + ((lane & 16) ? 16 : 0));
}

// ---------------------------------------------------------------------------
// Pack a row-major f32 weight [K x N] into bf16 B-fragment layout:
// fragment f = fk*(N/16)+fn holds tile (k0=fk*32, n0=fn*16); within a fragment
// lane l, elem e -> K = k0 + (l>=16?16:0) + e, N = n0 + (l&15).
__global__ __launch_bounds__(256) void pack_b_kernel(const float* __restrict__ W,
                                                     bf16* __restrict__ P, int K, int N) {
    int tid = blockIdx.x * 256 + threadIdx.x;
    if (tid >= K * N) return;
    int f = tid >> 9;
    int r = tid & 511;
    int lane = r >> 4, e = r & 15;
    int ntiles = N >> 4;
    int fk = f / ntiles, fn = f - fk * ntiles;
    int k = fk * 32 + ((lane >= 16) ? 16 : 0) + e;
    int n = fn * 16 + (lane & 15);
    P[tid] = (bf16)W[(size_t)k * N + n];
}

// f32 -> bf16 elementwise
__global__ __launch_bounds__(256) void f2b_kernel(const float* __restrict__ X,
                                                  bf16* __restrict__ Y, int n) {
    int i = blockIdx.x * 256 + threadIdx.x;
    if (i < n) Y[i] = (bf16)X[i];
}

// ---------------------------------------------------------------------------
// WMMA GEMM: C[M,N] = A[M,K](bf16,row-major) * Bp(packed bf16)
//   + bias[N]? + resid[M,N]?  -> optional relu -> outF (f32) / outB (bf16)
// Block 256 thr = 8 waves (2 Mx4 N); block tile 64x128; each wave computes a
// 32x32 micro-tile: 2 A frags x 2 B frags = 4 wmma per 32-K step.
// M multiple of 16; N multiple of 128; K multiple of 32.
__global__ __launch_bounds__(256) void gemm_bf16_kernel(
    const bf16* __restrict__ A, const bf16* __restrict__ Bp,
    const float* __restrict__ bias, const float* __restrict__ resid,
    float* __restrict__ outF, bf16* __restrict__ outB,
    int M, int N, int K, int relu)
{
    int lane  = threadIdx.x & 31;
    int w     = threadIdx.x >> 5;
    int mrow0 = blockIdx.y * 64 + (w & 1) * 32;
    if (mrow0 >= M) return;
    int mrow1 = mrow0 + 16;
    bool m1ok = (mrow1 < M);
    int n0    = blockIdx.x * 128 + (w >> 1) * 32;
    int ntiles = N >> 4;

    int arow0 = mrow0 + (lane & 15);
    int arow1 = (m1ok ? mrow1 : mrow0) + (lane & 15);
    int koff  = (lane & 16) ? 16 : 0;

    const bf16* ap0 = A + (size_t)arow0 * K + koff;
    const bf16* ap1 = A + (size_t)arow1 * K + koff;
    const bf16* bp0 = Bp + (size_t)(n0 >> 4) * 512 + lane * 16;
    const bf16* bp1 = bp0 + 512;
    size_t bstep = (size_t)ntiles * 512;

    v8f acc00 = {0.f,0.f,0.f,0.f,0.f,0.f,0.f,0.f};
    v8f acc01 = acc00, acc10 = acc00, acc11 = acc00;

    for (int kk = 0; kk < K; kk += 32) {
        __builtin_prefetch(bp0 + bstep, 0, 1);   // next K-step B tiles (speculative)
        v16bf a0 = a_frag_at(ap0);
        v16bf a1 = a_frag_at(ap1);
        v16bf b0 = *(const v16bf*)bp0;
        v16bf b1 = *(const v16bf*)bp1;
        acc00 = wmma_bf16(a0, b0, acc00);
        acc01 = wmma_bf16(a0, b1, acc01);
        acc10 = wmma_bf16(a1, b0, acc10);
        acc11 = wmma_bf16(a1, b1, acc11);
        ap0 += 32; ap1 += 32; bp0 += bstep; bp1 += bstep;
    }

    #pragma unroll
    for (int mi = 0; mi < 2; ++mi) {
        if (mi == 1 && !m1ok) break;
        int mb = mrow0 + mi * 16;
        #pragma unroll
        for (int j = 0; j < 2; ++j) {
            v8f acc = mi ? (j ? acc11 : acc10) : (j ? acc01 : acc00);
            int col = n0 + j * 16 + (lane & 15);
            float bv = bias ? bias[col] : 0.f;
            #pragma unroll
            for (int r = 0; r < 8; ++r) {
                int row = mb + r + ((lane & 16) ? 8 : 0);
                size_t o = (size_t)row * N + col;
                float v = acc[r] + bv;
                if (resid) v += resid[o];
                if (relu)  v = fmaxf(v, 0.f);
                if (outF)  outF[o] = v;
                if (outB)  outB[o] = (bf16)v;
            }
        }
    }
}

// ---------------------------------------------------------------------------
// LayerNorm over 256 cols, one wave per row; optional relu; f32/bf16 outs.
__global__ __launch_bounds__(256) void ln_kernel(
    const float* __restrict__ X, const float* __restrict__ g, const float* __restrict__ b,
    bf16* __restrict__ outB, float* __restrict__ outF, int M, int relu)
{
    int lane = threadIdx.x & 31;
    int w = threadIdx.x >> 5;
    int row = blockIdx.x * 8 + w;
    if (row >= M) return;
    const float* x = X + (size_t)row * DM;
    float v[8], s = 0.f, s2 = 0.f;
    #pragma unroll
    for (int t = 0; t < 8; ++t) { v[t] = x[lane + 32 * t]; s += v[t]; s2 += v[t] * v[t]; }
    #pragma unroll
    for (int m = 16; m; m >>= 1) { s += __shfl_xor(s, m); s2 += __shfl_xor(s2, m); }
    float mean = s * (1.f / DM);
    float var  = s2 * (1.f / DM) - mean * mean;
    float inv  = rsqrtf(var + 1e-5f);
    #pragma unroll
    for (int t = 0; t < 8; ++t) {
        int c = lane + 32 * t;
        float o = (v[t] - mean) * inv * g[c] + b[c];
        if (relu) o = fmaxf(o, 0.f);
        if (outB) outB[(size_t)row * DM + c] = (bf16)o;
        if (outF) outF[(size_t)row * DM + c] = o;
    }
}

// Sum of squares per 256-col row.
__global__ __launch_bounds__(256) void rowsq_kernel(const float* __restrict__ X,
                                                    float* __restrict__ out, int M) {
    int lane = threadIdx.x & 31;
    int w = threadIdx.x >> 5;
    int row = blockIdx.x * 8 + w;
    if (row >= M) return;
    float s = 0.f;
    #pragma unroll
    for (int t = 0; t < 8; ++t) { float v = X[(size_t)row * DM + lane + 32 * t]; s += v * v; }
    #pragma unroll
    for (int m = 16; m; m >>= 1) s += __shfl_xor(s, m);
    if (lane == 0) out[row] = s;
}

// ---------------------------------------------------------------------------
// kNN: 16 queries per block; score = 2*q.c - |c|^2 via WMMA; unordered top-96
// replace-min lists in LDS maintained by 16 owner lanes. Each wave covers two
// 16-candidate tiles per iteration (256 candidates per block iteration).
__global__ __launch_bounds__(256) void knn_topk_kernel(
    const bf16* __restrict__ Qb,   // [NB x 256] bf16 row-major
    const bf16* __restrict__ Cb,   // [NCAND x 256] bf16 row-major (acts as packed B)
    const float* __restrict__ C2,  // [NCAND]
    int ncand, int* __restrict__ tidx, float* __restrict__ tsc)
{
    __shared__ float sbuf[8][16][32];
    __shared__ float lval[16][CTX];
    __shared__ int   lidx[16][CTX];
    __shared__ int   lcnt[16];
    __shared__ float lmin[16];
    __shared__ int   lam[16];

    int lane = threadIdx.x & 31;
    int w    = threadIdx.x >> 5;
    int q0   = blockIdx.x * 16;
    if (threadIdx.x < 16) { lcnt[threadIdx.x] = 0; lmin[threadIdx.x] = -3.0e38f; lam[threadIdx.x] = 0; }
    __syncthreads();

    v16bf afrag[8];
    #pragma unroll
    for (int t = 0; t < 8; ++t) afrag[t] = load_a_frag(Qb, DM, q0 + (lane & 15), t * 32, lane);

    bool owner = (lane == 0) || (lane == 16);
    int  myq   = w * 2 + (lane >> 4);

    for (int base = 0; base < ncand; base += 256) {
        #pragma unroll
        for (int s = 0; s < 2; ++s) {
            int cb = base + w * 32 + s * 16;
            int ci = cb + (lane & 15);
            int crow = ci < ncand ? ci : (ncand - 1);
            const bf16* cp = Cb + (size_t)crow * DM + ((lane & 16) ? 16 : 0);
            v8f acc = {0.f,0.f,0.f,0.f,0.f,0.f,0.f,0.f};
            #pragma unroll
            for (int t = 0; t < 8; ++t) {
                v16bf bfr = *(const v16bf*)(cp + t * 32);
                acc = wmma_bf16(afrag[t], bfr, acc);
            }
            float c2v = C2[crow];
            bool valid = ci < ncand;
            #pragma unroll
            for (int r = 0; r < 8; ++r) {
                int q = r + ((lane & 16) ? 8 : 0);
                sbuf[w][q][s * 16 + (lane & 15)] = valid ? (2.f * acc[r] - c2v) : -3.0e38f;
            }
        }
        __syncthreads();
        if (owner) {
            int cnt = lcnt[myq];
            float mn = lmin[myq];
            int   am = lam[myq];
            for (int wv = 0; wv < 8; ++wv) {
                for (int c = 0; c < 32; ++c) {
                    float s = sbuf[wv][myq][c];
                    int   cand = base + wv * 32 + c;
                    if (cnt < CTX) {
                        lval[myq][cnt] = s; lidx[myq][cnt] = cand; ++cnt;
                        if (cnt == CTX) {
                            mn = lval[myq][0]; am = 0;
                            for (int j = 1; j < CTX; ++j)
                                if (lval[myq][j] < mn) { mn = lval[myq][j]; am = j; }
                        }
                    } else if (s > mn) {
                        lval[myq][am] = s; lidx[myq][am] = cand;
                        mn = lval[myq][0]; am = 0;
                        for (int j = 1; j < CTX; ++j)
                            if (lval[myq][j] < mn) { mn = lval[myq][j]; am = j; }
                    }
                }
            }
            lcnt[myq] = cnt; lmin[myq] = mn; lam[myq] = am;
        }
        __syncthreads();
    }
    if (owner) {
        for (int j = 0; j < CTX; ++j) {
            tsc[(size_t)(q0 + myq) * CTX + j]  = lval[myq][j];
            tidx[(size_t)(q0 + myq) * CTX + j] = lidx[myq][j];
        }
    }
}

// Softmax over the 96 context scores; sim = score - |k|^2. One wave per query.
__global__ __launch_bounds__(256) void softmax_kernel(const float* __restrict__ tsc,
                                                      const float* __restrict__ Q2,
                                                      float* __restrict__ probs) {
    int lane = threadIdx.x & 31;
    int w = threadIdx.x >> 5;
    int q = blockIdx.x * 8 + w;
    if (q >= NB) return;
    float k2 = Q2[q];
    float s[3];
    #pragma unroll
    for (int t = 0; t < 3; ++t) s[t] = tsc[(size_t)q * CTX + lane + 32 * t] - k2;
    float mx = fmaxf(s[0], fmaxf(s[1], s[2]));
    #pragma unroll
    for (int m = 16; m; m >>= 1) mx = fmaxf(mx, __shfl_xor(mx, m));
    float e[3], sum = 0.f;
    #pragma unroll
    for (int t = 0; t < 3; ++t) { e[t] = __expf(s[t] - mx); sum += e[t]; }
    #pragma unroll
    for (int m = 16; m; m >>= 1) sum += __shfl_xor(sum, m);
    float inv = 1.f / sum;
    #pragma unroll
    for (int t = 0; t < 3; ++t) probs[(size_t)q * CTX + lane + 32 * t] = e[t] * inv;
}

// Build Diff rows: D[(b,c), :] = k[b] - cand_k[idx[b,c]]  (bf16), for QCH queries.
__global__ __launch_bounds__(256) void build_diff_kernel(
    const float* __restrict__ xk, const float* __restrict__ CK,
    const int* __restrict__ tidx, int qbase, bf16* __restrict__ D) {
    int tid = blockIdx.x * 256 + threadIdx.x;     // QCH*CTX*256 threads
    int d  = tid & 255;
    int rr = tid >> 8;
    int b = qbase + rr / CTX;
    int c = rr - (rr / CTX) * CTX;
    int ci = tidx[(size_t)b * CTX + c];
    D[tid] = (bf16)(xk[(size_t)b * DM + d] - CK[(size_t)ci * DM + d]);
}

// h[b,:] += sum_c probs[b,c] * (label_emb[y_c,:] + V[(b,c),:])
__global__ __launch_bounds__(256) void ctx_reduce_kernel(
    const float* __restrict__ probs, const bf16* __restrict__ Vb,
    const int* __restrict__ tidx, const int* __restrict__ cy,
    const float* __restrict__ lemb, float* __restrict__ h, int qbase) {
    __shared__ float p[CTX];
    __shared__ int   y[CTX];
    int b = qbase + blockIdx.x;
    int d = threadIdx.x;
    if (threadIdx.x < CTX) {
        p[threadIdx.x] = probs[(size_t)b * CTX + threadIdx.x];
        y[threadIdx.x] = cy[tidx[(size_t)b * CTX + threadIdx.x]];
    }
    __syncthreads();
    float acc = 0.f;
    size_t rbase = (size_t)blockIdx.x * CTX;
    for (int c = 0; c < CTX; ++c) {
        float v = (float)Vb[(rbase + c) * DM + d] + lemb[(size_t)y[c] * DM + d];
        acc += p[c] * v;
    }
    h[(size_t)b * DM + d] += acc;
}

// out[b,o] = sum_d Xr[b,d]*W[d,o] + bias[o]   (Xr already relu(LN(h)))
__global__ __launch_bounds__(256) void head_kernel(const float* __restrict__ Xr,
                                                   const float* __restrict__ W,
                                                   const float* __restrict__ bias,
                                                   float* __restrict__ out) {
    int t = blockIdx.x * 256 + threadIdx.x;
    if (t >= NB * NOUT) return;
    int b = t / NOUT, o = t - b * NOUT;
    float acc = bias[o];
    for (int d = 0; d < DM; ++d) acc += Xr[(size_t)b * DM + d] * W[d * NOUT + o];
    out[t] = acc;
}

// ===========================================================================
extern "C" void kernel_launch(void* const* d_in, const int* in_sizes, int n_in,
                              void* d_out, int out_size, void* d_ws, size_t ws_size,
                              hipStream_t stream) {
    const float* x_in      = (const float*)d_in[0];
    const float* cand_x    = (const float*)d_in[1];
    const int*   cand_y    = (const int*)d_in[2];
    const float* W_lin     = (const float*)d_in[4];
    const float* b_lin     = (const float*)d_in[5];
    const float* e0W1      = (const float*)d_in[6];
    const float* e0b1      = (const float*)d_in[7];
    const float* e0W2      = (const float*)d_in[8];
    const float* e0b2      = (const float*)d_in[9];
    const float* e1g       = (const float*)d_in[10];
    const float* e1b       = (const float*)d_in[11];
    const float* e1W1      = (const float*)d_in[12];
    const float* e1b1      = (const float*)d_in[13];
    const float* e1W2      = (const float*)d_in[14];
    const float* e1b2      = (const float*)d_in[15];
    const float* mixg      = (const float*)d_in[16];
    const float* mixb      = (const float*)d_in[17];
    const float* K_W       = (const float*)d_in[18];
    const float* K_b       = (const float*)d_in[19];
    const float* lemb      = (const float*)d_in[20];
    const float* T_W1      = (const float*)d_in[21];
    const float* T_b1      = (const float*)d_in[22];
    const float* T_W2      = (const float*)d_in[23];
    const float* p0g       = (const float*)d_in[24];
    const float* p0b       = (const float*)d_in[25];
    const float* p0W1      = (const float*)d_in[26];
    const float* p0b1      = (const float*)d_in[27];
    const float* p0W2      = (const float*)d_in[28];
    const float* p0b2      = (const float*)d_in[29];
    const float* p1g       = (const float*)d_in[30];
    const float* p1b       = (const float*)d_in[31];
    const float* p1W1      = (const float*)d_in[32];
    const float* p1b1      = (const float*)d_in[33];
    const float* p1W2      = (const float*)d_in[34];
    const float* p1b2      = (const float*)d_in[35];
    const float* headg     = (const float*)d_in[36];
    const float* headb     = (const float*)d_in[37];
    const float* head_W    = (const float*)d_in[38];
    const float* head_bias = (const float*)d_in[39];
    float* out = (float*)d_out;

    // ---- workspace layout ----
    char* base = (char*)d_ws;
    size_t off = 0;
    auto alloc = [&](size_t bytes) { size_t o = off; off = (off + bytes + 255) & ~(size_t)255; return o; };

    // packed bf16 weights
    size_t o_pwlin  = alloc((size_t)DIN * DM * 2);
    size_t o_pe0w1  = alloc((size_t)DM * DB * 2);
    size_t o_pe0w2  = alloc((size_t)DB * DM * 2);
    size_t o_pe1w1  = alloc((size_t)DM * DB * 2);
    size_t o_pe1w2  = alloc((size_t)DB * DM * 2);
    size_t o_pkw    = alloc((size_t)DM * DM * 2);
    size_t o_ptw1   = alloc((size_t)DM * DB * 2);
    size_t o_ptw2   = alloc((size_t)DB * DM * 2);
    size_t o_pp0w1  = alloc((size_t)DM * DB * 2);
    size_t o_pp0w2  = alloc((size_t)DB * DM * 2);
    size_t o_pp1w1  = alloc((size_t)DM * DB * 2);
    size_t o_pp1w2  = alloc((size_t)DB * DM * 2);
    // persistent candidate keys
    size_t o_CK   = alloc((size_t)NCAND * DM * 4);
    size_t o_CKb  = alloc((size_t)NCAND * DM * 2);
    size_t o_C2   = alloc((size_t)NCAND * 4);
    size_t o_Q2   = alloc((size_t)NB * 4);
    // query buffers
    size_t o_xinb = alloc((size_t)NB * DIN * 2);
    size_t o_xh0  = alloc((size_t)NB * DM * 4);
    size_t o_xh1  = alloc((size_t)NB * DM * 4);
    size_t o_xb   = alloc((size_t)NB * DM * 2);
    size_t o_xt   = alloc((size_t)NB * DB * 2);
    size_t o_xk   = alloc((size_t)NB * DM * 4);
    size_t o_xkb  = alloc((size_t)NB * DM * 2);
    // top-k
    size_t o_tidx = alloc((size_t)NB * CTX * 4);
    size_t o_tsc  = alloc((size_t)NB * CTX * 4);
    size_t o_prob = alloc((size_t)NB * CTX * 4);
    // shared POOL: candidate-encode chunk buffers / ctx chunk buffers
    size_t poolA = off;
    size_t o_cin = alloc((size_t)CCH * DIN * 2);
    size_t o_cA  = alloc((size_t)CCH * DM * 4);
    size_t o_cC  = alloc((size_t)CCH * DM * 4);
    size_t o_cBb = alloc((size_t)CCH * DM * 2);
    size_t o_cT  = alloc((size_t)CCH * DB * 2);
    size_t encEnd = off;
    off = poolA;                                   // overlay ctx buffers on pool
    size_t o_diff = alloc((size_t)QCH * CTX * DM * 2);
    size_t o_U    = alloc((size_t)QCH * CTX * DB * 2);
    size_t o_Vb   = alloc((size_t)QCH * CTX * DM * 2);
    if (off < encEnd) off = encEnd;
    (void)ws_size;

    bf16*  pwlin = (bf16*)(base + o_pwlin);
    bf16*  pe0w1 = (bf16*)(base + o_pe0w1);
    bf16*  pe0w2 = (bf16*)(base + o_pe0w2);
    bf16*  pe1w1 = (bf16*)(base + o_pe1w1);
    bf16*  pe1w2 = (bf16*)(base + o_pe1w2);
    bf16*  pkw   = (bf16*)(base + o_pkw);
    bf16*  ptw1  = (bf16*)(base + o_ptw1);
    bf16*  ptw2  = (bf16*)(base + o_ptw2);
    bf16*  pp0w1 = (bf16*)(base + o_pp0w1);
    bf16*  pp0w2 = (bf16*)(base + o_pp0w2);
    bf16*  pp1w1 = (bf16*)(base + o_pp1w1);
    bf16*  pp1w2 = (bf16*)(base + o_pp1w2);
    float* CK  = (float*)(base + o_CK);
    bf16*  CKb = (bf16*)(base + o_CKb);
    float* C2  = (float*)(base + o_C2);
    float* Q2  = (float*)(base + o_Q2);
    bf16*  xinb= (bf16*)(base + o_xinb);
    float* xh0 = (float*)(base + o_xh0);
    float* xh1 = (float*)(base + o_xh1);
    bf16*  xb  = (bf16*)(base + o_xb);
    bf16*  xt  = (bf16*)(base + o_xt);
    float* xk  = (float*)(base + o_xk);
    bf16*  xkb = (bf16*)(base + o_xkb);
    int*   tidx= (int*)(base + o_tidx);
    float* tsc = (float*)(base + o_tsc);
    float* prob= (float*)(base + o_prob);
    bf16*  cin = (bf16*)(base + o_cin);
    float* cA  = (float*)(base + o_cA);
    float* cC  = (float*)(base + o_cC);
    bf16*  cBb = (bf16*)(base + o_cBb);
    bf16*  cT  = (bf16*)(base + o_cT);
    bf16*  Dif = (bf16*)(base + o_diff);
    bf16*  U   = (bf16*)(base + o_U);
    bf16*  Vb  = (bf16*)(base + o_Vb);

    auto pack = [&](const float* W, bf16* P, int K, int N) {
        int tot = K * N;
        pack_b_kernel<<<(tot + 255) / 256, 256, 0, stream>>>(W, P, K, N);
    };
    auto gemm = [&](const bf16* A, const bf16* Bp, const float* bias, const float* resid,
                    float* oF, bf16* oB, int M, int N, int K, int relu) {
        dim3 grid(N / 128, (M + 63) / 64);
        gemm_bf16_kernel<<<grid, 256, 0, stream>>>(A, Bp, bias, resid, oF, oB, M, N, K, relu);
    };
    auto lnorm = [&](const float* X, const float* g, const float* b,
                     bf16* oB, float* oF, int M, int relu) {
        ln_kernel<<<(M + 7) / 8, 256, 0, stream>>>(X, g, b, oB, oF, M, relu);
    };

    // 1) pack weights
    pack(W_lin, pwlin, DIN, DM);
    pack(e0W1, pe0w1, DM, DB);  pack(e0W2, pe0w2, DB, DM);
    pack(e1W1, pe1w1, DM, DB);  pack(e1W2, pe1w2, DB, DM);
    pack(K_W,  pkw,   DM, DM);
    pack(T_W1, ptw1,  DM, DB);  pack(T_W2, ptw2,  DB, DM);
    pack(p0W1, pp0w1, DM, DB);  pack(p0W2, pp0w2, DB, DM);
    pack(p1W1, pp1w1, DM, DB);  pack(p1W2, pp1w2, DB, DM);

    // 2) encode candidates in chunks -> CK / CKb
    for (int ch = 0; ch < NCAND; ch += CCH) {
        int n = CCH * DIN;
        f2b_kernel<<<(n + 255) / 256, 256, 0, stream>>>(cand_x + (size_t)ch * DIN, cin, n);
        gemm(cin, pwlin, b_lin, nullptr, cA, cBb, CCH, DM, DIN, 0);
        gemm(cBb, pe0w1, e0b1, nullptr, nullptr, cT, CCH, DB, DM, 1);
        gemm(cT,  pe0w2, e0b2, cA, cC, nullptr, CCH, DM, DB, 0);
        lnorm(cC, e1g, e1b, cBb, nullptr, CCH, 0);
        gemm(cBb, pe1w1, e1b1, nullptr, nullptr, cT, CCH, DB, DM, 1);
        gemm(cT,  pe1w2, e1b2, cC, cA, nullptr, CCH, DM, DB, 0);
        lnorm(cA, mixg, mixb, cBb, nullptr, CCH, 0);
        gemm(cBb, pkw, K_b, nullptr, CK + (size_t)ch * DM, CKb + (size_t)ch * DM, CCH, DM, DM, 0);
    }
    rowsq_kernel<<<(NCAND + 7) / 8, 256, 0, stream>>>(CK, C2, NCAND);

    // 3) encode queries -> xh0 (h), xk/xkb (k)
    {
        int n = NB * DIN;
        f2b_kernel<<<(n + 255) / 256, 256, 0, stream>>>(x_in, xinb, n);
        gemm(xinb, pwlin, b_lin, nullptr, xh0, xb, NB, DM, DIN, 0);
        gemm(xb, pe0w1, e0b1, nullptr, nullptr, xt, NB, DB, DM, 1);
        gemm(xt, pe0w2, e0b2, xh0, xh1, nullptr, NB, DM, DB, 0);
        lnorm(xh1, e1g, e1b, xb, nullptr, NB, 0);
        gemm(xb, pe1w1, e1b1, nullptr, nullptr, xt, NB, DB, DM, 1);
        gemm(xt, pe1w2, e1b2, xh1, xh0, nullptr, NB, DM, DB, 0);   // final h -> xh0
        lnorm(xh0, mixg, mixb, xb, nullptr, NB, 0);
        gemm(xb, pkw, K_b, nullptr, xk, xkb, NB, DM, DM, 0);
    }
    rowsq_kernel<<<(NB + 7) / 8, 256, 0, stream>>>(xk, Q2, NB);

    // 4) kNN top-96 + softmax
    knn_topk_kernel<<<NB / 16, 256, 0, stream>>>(xkb, CKb, C2, NCAND, tidx, tsc);
    softmax_kernel<<<NB / 8, 256, 0, stream>>>(tsc, Q2, prob);

    // 5) context value MLP + weighted reduction into h (xh0), chunked
    for (int qb = 0; qb < NB; qb += QCH) {
        int rows = QCH * CTX;
        build_diff_kernel<<<rows, 256, 0, stream>>>(xk, CK, tidx, qb, Dif);
        gemm(Dif, ptw1, T_b1, nullptr, nullptr, U, rows, DB, DM, 1);
        gemm(U,   ptw2, nullptr, nullptr, nullptr, Vb, rows, DM, DB, 0);
        ctx_reduce_kernel<<<QCH, 256, 0, stream>>>(prob, Vb, tidx, cand_y, lemb, xh0, qb);
    }

    // 6) post blocks p0, p1
    lnorm(xh0, p0g, p0b, xb, nullptr, NB, 0);
    gemm(xb, pp0w1, p0b1, nullptr, nullptr, xt, NB, DB, DM, 1);
    gemm(xt, pp0w2, p0b2, xh0, xh1, nullptr, NB, DM, DB, 0);
    lnorm(xh1, p1g, p1b, xb, nullptr, NB, 0);
    gemm(xb, pp1w1, p1b1, nullptr, nullptr, xt, NB, DB, DM, 1);
    gemm(xt, pp1w2, p1b2, xh1, xh0, nullptr, NB, DM, DB, 0);

    // 7) head: relu(LN(h)) @ head_W + head_bias
    lnorm(xh0, headg, headb, nullptr, xh1, NB, 1);
    head_kernel<<<(NB * NOUT + 255) / 256, 256, 0, stream>>>(xh1, head_W, head_bias, out);
}
